// TitanLLaMADecoderLayer_704374636635
// MI455X (gfx1250) — matrix operand order repaired and checked by reading, not compiled, not run on hardware
//
#include <hip/hip_runtime.h>

typedef __attribute__((ext_vector_type(16))) __bf16 v16bf;
typedef __attribute__((ext_vector_type(8)))  float  v8f;

// ---------------- constants ----------------
#define S_   4128
#define D_   2048
#define H_   32
#define DH_  64
#define HD_  2048        // H*DH
#define QN_  6144        // 3*H*DH
#define FF_  11008
#define SEG_ 516         // N = SEG+LM
#define P_   4
#define W_   8           // S/N
#define QT_  33          // ceil(516/16)

// ---------------- CDNA5 async global->LDS (guarded; fallback = direct copy) ----
#if defined(__has_builtin)
#  if __has_builtin(__builtin_amdgcn_global_load_async_to_lds_b128)
#    define HAVE_ASYNC_LDS 1
#  endif
#  if __has_builtin(__builtin_amdgcn_s_wait_asynccnt)
#    define HAVE_WAIT_ASYNC 1
#  endif
#endif
#ifndef HAVE_ASYNC_LDS
#  define HAVE_ASYNC_LDS 0
#endif
#ifndef HAVE_WAIT_ASYNC
#  define HAVE_WAIT_ASYNC 0
#endif

typedef int v4i_vs __attribute__((vector_size(16)));  // matches builtin param pointee

__device__ __forceinline__ void async_copy_b128(const unsigned short* g, unsigned short* l) {
#if HAVE_ASYNC_LDS
    __builtin_amdgcn_global_load_async_to_lds_b128(
        (__attribute__((address_space(1))) v4i_vs*)g,
        (__attribute__((address_space(3))) v4i_vs*)l, 0, 0);
#else
    *(uint4*)l = *(const uint4*)g;
#endif
}

__device__ __forceinline__ void wait_async_lds() {
#if HAVE_ASYNC_LDS
#  if HAVE_WAIT_ASYNC
    __builtin_amdgcn_s_wait_asynccnt(0);
#  else
    asm volatile("s_wait_asynccnt 0x0" ::: "memory");
#  endif
#endif
}

// ---------------- helpers ----------------
__device__ __forceinline__ unsigned short f2bf(float f) {
    union { float f; unsigned int u; } a; a.f = f;
    unsigned int r = a.u + 0x7FFFu + ((a.u >> 16) & 1u);
    return (unsigned short)(r >> 16);
}

__device__ __forceinline__ v8f wmma_bf16(v16bf a, v16bf b, v8f c) {
    return __builtin_amdgcn_wmma_f32_16x16x32_bf16(false, a, false, b, (short)0, c, false, false);
}

// A fragment (16x32 bf16): lane -> K [hi8..hi8+7],[16+hi8..+7] of row (lane&15)
__device__ __forceinline__ v16bf frag_a_from_row(const unsigned short* rowp, int koff, int lane) {
    int hi = (lane >> 4) & 1;
    const unsigned short* p = rowp + koff + hi * 8;
    union { v16bf v; uint4 q[2]; } u;
    u.q[0] = *(const uint4*)(p);
    u.q[1] = *(const uint4*)(p + 16);
    return u.v;
}

// B fragment (32x16 bf16) from K-contiguous row (N = lane&15 selects row pointer):
// lane -> K [hi8..hi8+7],[16+hi8..+7]
__device__ __forceinline__ v16bf frag_b_from_row(const unsigned short* rowp, int koff, int lane) {
    int hi = (lane >> 4) & 1;
    const unsigned short* p = rowp + koff + hi * 8;
    union { v16bf v; uint4 q[2]; } u;
    u.q[0] = *(const uint4*)(p);
    u.q[1] = *(const uint4*)(p + 16);
    return u.v;
}

// ---------------- fp32 -> bf16 convert ----------------
__global__ void cvt_bf16_kernel(const float* __restrict__ in, unsigned short* __restrict__ out, size_t n) {
    size_t i = (size_t)blockIdx.x * blockDim.x + threadIdx.x;
    size_t stride = (size_t)gridDim.x * blockDim.x;
    for (; i < n; i += stride) out[i] = f2bf(in[i]);
}

// ---------------- fused double RMSNorm (input_ln then attn_norm) -> bf16 ----------------
__global__ __launch_bounds__(256) void rmsnorm2_kernel(
    const float* __restrict__ x, const float* __restrict__ w1,
    const float* __restrict__ w2, unsigned short* __restrict__ out, int cols) {
    __shared__ float red[256];
    int row = blockIdx.x, tid = threadIdx.x;
    const float* xr = x + (size_t)row * cols;
    float v[8]; float ss = 0.f;
#pragma unroll
    for (int i = 0; i < 8; i++) { v[i] = xr[tid + i * 256]; ss += v[i] * v[i]; }
    red[tid] = ss; __syncthreads();
    for (int s = 128; s > 0; s >>= 1) { if (tid < s) red[tid] += red[tid + s]; __syncthreads(); }
    float r1 = rsqrtf(red[0] / (float)cols + 1e-6f);
    __syncthreads();
    float hb[8]; float ss2 = 0.f;
#pragma unroll
    for (int i = 0; i < 8; i++) { hb[i] = w1[tid + i * 256] * v[i] * r1; ss2 += hb[i] * hb[i]; }
    red[tid] = ss2; __syncthreads();
    for (int s = 128; s > 0; s >>= 1) { if (tid < s) red[tid] += red[tid + s]; __syncthreads(); }
    float r2 = rsqrtf(red[0] / (float)cols + 1e-6f);
#pragma unroll
    for (int i = 0; i < 8; i++)
        out[(size_t)row * cols + tid + i * 256] = f2bf(w2[tid + i * 256] * hb[i] * r2);
}

// ---------------- single RMSNorm -> bf16 ----------------
__global__ __launch_bounds__(256) void rmsnorm1_kernel(
    const float* __restrict__ x, const float* __restrict__ w,
    unsigned short* __restrict__ out, int cols) {
    __shared__ float red[256];
    int row = blockIdx.x, tid = threadIdx.x;
    const float* xr = x + (size_t)row * cols;
    float v[8]; float ss = 0.f;
#pragma unroll
    for (int i = 0; i < 8; i++) { v[i] = xr[tid + i * 256]; ss += v[i] * v[i]; }
    red[tid] = ss; __syncthreads();
    for (int s = 128; s > 0; s >>= 1) { if (tid < s) red[tid] += red[tid + s]; __syncthreads(); }
    float r1 = rsqrtf(red[0] / (float)cols + 1e-6f);
#pragma unroll
    for (int i = 0; i < 8; i++)
        out[(size_t)row * cols + tid + i * 256] = f2bf(w[tid + i * 256] * v[i] * r1);
}

// ---------------- WMMA GEMM: C[M,N] = A[M,K]@B[K,N], bf16 in, fp32 acc ----------------
// MODE 0: store f32   MODE 1: store f32 + residual   MODE 2: dual-B, silu(C1)*C2 -> bf16
// TBN: block N-tile (128 for single-B, 64 for dual-B). 8 waves: 4 (m) x 2 (n).
// Double-buffered LDS; A tile staged via CDNA5 async global->LDS, B staged transposed
// through registers (scatter). One barrier per k-step.
#define BM 128
#define BK 32
template <int MODE, int TBN>
__global__ __launch_bounds__(256) void gemm_bf16_kernel(
    const unsigned short* __restrict__ A, const unsigned short* __restrict__ B,
    const unsigned short* __restrict__ B2, float* __restrict__ Cf,
    unsigned short* __restrict__ Cbf, const float* __restrict__ Res,
    int M, int N, int K) {
    constexpr int NT = TBN / 32;      // 16-wide n-tiles per wave
    constexpr int BSEG = TBN / 8;     // B elems staged per thread
    __shared__ unsigned short As[2][BM * BK];
    __shared__ unsigned short Bt[2][TBN * BK];
    __shared__ unsigned short Bt2[2][(MODE == 2) ? TBN * BK : 8];

    int tid = threadIdx.x, lane = tid & 31, wid = tid >> 5;
    int wm = wid >> 1, wn = wid & 1;
    int m0 = blockIdx.y * BM, n0 = blockIdx.x * TBN;

    v8f zero = {0.f, 0.f, 0.f, 0.f, 0.f, 0.f, 0.f, 0.f};
    v8f acc[2][NT], acc2[2][NT];
#pragma unroll
    for (int i = 0; i < 2; i++)
#pragma unroll
        for (int j = 0; j < NT; j++) { acc[i][j] = zero; acc2[i][j] = zero; }

    // per-thread staging assignment
    const int ra = tid >> 1, ksa = (tid & 1) * 16;      // A: 16 elems at (ra, ksa)
    const int kb = tid >> 3, cb = (tid & 7) * BSEG;     // B: BSEG elems at (kb, cb)
    const int gra = m0 + ra;
    const unsigned short* Arow = A + (size_t)gra * K + ksa;
    const unsigned short* Brow = B + (size_t)kb * N + n0 + cb;
    const unsigned short* B2row = (MODE == 2) ? (B2 + (size_t)kb * N + n0 + cb) : nullptr;

    uint4 bR0, bR1, b2R0, b2R1;
    auto issueA = [&](int kt, int buf) {
        if (gra < M) {
            async_copy_b128(Arow + kt, &As[buf][ra * BK + ksa]);
            async_copy_b128(Arow + kt + 8, &As[buf][ra * BK + ksa + 8]);
        }
    };
    auto loadB = [&](int kt) {
        bR0 = *(const uint4*)(Brow + (size_t)kt * N);
        if (BSEG == 16) bR1 = *(const uint4*)(Brow + (size_t)kt * N + 8);
        if (MODE == 2) {
            b2R0 = *(const uint4*)(B2row + (size_t)kt * N);
            if (BSEG == 16) b2R1 = *(const uint4*)(B2row + (size_t)kt * N + 8);
        }
    };
    auto commitB = [&](int buf) {
        uint4 btmp[2] = {bR0, bR1};
        const unsigned short* bs = (const unsigned short*)btmp;
#pragma unroll
        for (int j = 0; j < BSEG; j++) Bt[buf][(cb + j) * BK + kb] = bs[j];
        if (MODE == 2) {
            uint4 b2tmp[2] = {b2R0, b2R1};
            const unsigned short* b2s = (const unsigned short*)b2tmp;
#pragma unroll
            for (int j = 0; j < BSEG; j++) Bt2[buf][(cb + j) * BK + kb] = b2s[j];
        }
    };
    auto compute = [&](int buf) {
        v16bf af[2], bf[NT], bf2[NT];
#pragma unroll
        for (int tm = 0; tm < 2; tm++)
            af[tm] = frag_a_from_row(&As[buf][(wm * 32 + tm * 16 + (lane & 15)) * BK], 0, lane);
#pragma unroll
        for (int tn = 0; tn < NT; tn++)
            bf[tn] = frag_b_from_row(&Bt[buf][(wn * (TBN / 2) + tn * 16 + (lane & 15)) * BK], 0, lane);
        if (MODE == 2) {
#pragma unroll
            for (int tn = 0; tn < NT; tn++)
                bf2[tn] = frag_b_from_row(&Bt2[buf][(wn * (TBN / 2) + tn * 16 + (lane & 15)) * BK], 0, lane);
        }
#pragma unroll
        for (int tm = 0; tm < 2; tm++)
#pragma unroll
            for (int tn = 0; tn < NT; tn++) {
                acc[tm][tn] = wmma_bf16(af[tm], bf[tn], acc[tm][tn]);
                if (MODE == 2) acc2[tm][tn] = wmma_bf16(af[tm], bf2[tn], acc2[tm][tn]);
            }
    };

    // pipelined main loop: tile k+1 in flight (async A -> LDS, B -> regs) while WMMA on k
    issueA(0, 0);
    loadB(0);
    wait_async_lds();
    commitB(0);
    __syncthreads();
    int buf = 0;
    for (int kt = BK; kt < K; kt += BK) {
        issueA(kt, buf ^ 1);
        loadB(kt);
        if (kt + 2 * BK < K) {
            __builtin_prefetch(Arow + kt + 2 * BK, 0, 0);
            __builtin_prefetch(Brow + (size_t)(kt + 2 * BK) * N, 0, 0);
        }
        compute(buf);
        wait_async_lds();
        commitB(buf ^ 1);
        __syncthreads();
        buf ^= 1;
    }
    compute(buf);

    // epilogue
    int hi = (lane >> 4) & 1;
#pragma unroll
    for (int tm = 0; tm < 2; tm++)
#pragma unroll
        for (int tn = 0; tn < NT; tn++) {
            int rbase = m0 + wm * 32 + tm * 16 + hi * 8;
            int col = n0 + wn * (TBN / 2) + tn * 16 + (lane & 15);
#pragma unroll
            for (int r = 0; r < 8; r++) {
                int row = rbase + r;
                if (row < M && col < N) {
                    float v = acc[tm][tn][r];
                    if (MODE == 0) {
                        Cf[(size_t)row * N + col] = v;
                    } else if (MODE == 1) {
                        Cf[(size_t)row * N + col] = Res[(size_t)row * N + col] + v;
                    } else {
                        float g = v, u = acc2[tm][tn][r];
                        float s = g / (1.f + __expf(-g));
                        Cbf[(size_t)row * N + col] = f2bf(s * u);
                    }
                }
            }
        }
}

// ---------------- rotary + scale + bf16 repack to [H][S][64] ----------------
__global__ __launch_bounds__(256) void rotary_kernel(
    const float* __restrict__ qkv, unsigned short* __restrict__ qb,
    unsigned short* __restrict__ kb, unsigned short* __restrict__ vb, int S) {
    int s = blockIdx.x;
    for (int idx = threadIdx.x; idx < H_ * 32; idx += 256) {
        int h = idx >> 5, p = idx & 31;
        const float* base = qkv + (size_t)s * QN_;
        const float* q = base + h * DH_;
        const float* k = base + HD_ + h * DH_;
        const float* v = base + 2 * HD_ + h * DH_;
        float inv = __powf(10000.f, -(float)p / 32.f);
        float f = (float)s * inv;
        float sn, c; __sincosf(f, &sn, &c);
        size_t o = ((size_t)h * S + s) * DH_ + 2 * p;
        float q1 = q[2 * p], q2 = q[2 * p + 1];
        qb[o]     = f2bf((q1 * c - q2 * sn) * 0.125f);
        qb[o + 1] = f2bf((q2 * c + q1 * sn) * 0.125f);
        float k1 = k[2 * p], k2 = k[2 * p + 1];
        kb[o]     = f2bf(k1 * c - k2 * sn);
        kb[o + 1] = f2bf(k2 * c + k1 * sn);
        vb[o]     = f2bf(v[2 * p]);
        vb[o + 1] = f2bf(v[2 * p + 1]);
    }
}

// ---------------- flash attention, one wave per 16-query tile ----------------
__device__ __forceinline__ const unsigned short* key_row(
    const unsigned short* tok, const unsigned short* pm, int h, int w, int j, int S) {
    if (j < P_) return pm + ((size_t)h * P_ + j) * DH_;
    int t = j - P_;
    if (t >= 2 * SEG_) t = 2 * SEG_ - 1;
    int g = w * SEG_ + t - SEG_;
    if (g < 0) g = 0;
    if (g >= S) g = S - 1;
    return tok + ((size_t)h * S + g) * DH_;
}

__global__ __launch_bounds__(32) void attn_kernel(
    const unsigned short* __restrict__ qb, const unsigned short* __restrict__ kb,
    const unsigned short* __restrict__ vb, const unsigned short* __restrict__ pk,
    const unsigned short* __restrict__ pv, unsigned short* __restrict__ outb, int S) {
    __shared__ unsigned short p_lds[16 * 32];
    __shared__ unsigned short vt[64 * 32];
    int lane = threadIdx.x;
    int bid = blockIdx.x;
    int qt = bid % QT_;
    int w = (bid / QT_) % W_;
    int h = bid / (QT_ * W_);
    int hi = (lane >> 4) & 1;

    int iq = qt * 16 + (lane & 15);
    int iqc = iq < SEG_ ? iq : SEG_ - 1;
    const unsigned short* qrow = qb + ((size_t)h * S + (size_t)w * SEG_ + iqc) * DH_;
    v16bf aq0 = frag_a_from_row(qrow, 0, lane);
    v16bf aq1 = frag_a_from_row(qrow, 32, lane);

    v8f zero = {0.f, 0.f, 0.f, 0.f, 0.f, 0.f, 0.f, 0.f};
    float mstate[8], lstate[8];
    v8f o[4];
#pragma unroll
    for (int r = 0; r < 8; r++) { mstate[r] = -3.0e38f; lstate[r] = 0.f; }
#pragma unroll
    for (int t = 0; t < 4; t++) o[t] = zero;

    for (int j0 = 0; j0 < 1056; j0 += 32) {  // ceil(1036/32)*32 keys, padded+masked
        v8f sc[2];
#pragma unroll
        for (int nt = 0; nt < 2; nt++) {
            int j = j0 + nt * 16 + (lane & 15);
            const unsigned short* kr = key_row(kb, pk, h, w, j, S);
            v16bf b0 = frag_b_from_row(kr, 0, lane);
            v16bf b1 = frag_b_from_row(kr, 32, lane);
            v8f s = zero;
            s = wmma_bf16(aq0, b0, s);
            s = wmma_bf16(aq1, b1, s);
            sc[nt] = s;
        }
        // mask
#pragma unroll
        for (int nt = 0; nt < 2; nt++) {
            int j = j0 + nt * 16 + (lane & 15);
#pragma unroll
            for (int r = 0; r < 8; r++) {
                int i = qt * 16 + hi * 8 + r;
                bool vis;
                if (j < P_) vis = (i < SEG_);
                else {
                    int t = j - P_;
                    vis = (i < SEG_) && (t < 2 * SEG_) && (t >= i) && (t <= i + SEG_) &&
                          (w * SEG_ + t - SEG_ >= 0);
                }
                if (!vis) sc[nt][r] = -3.0e38f;
            }
        }
        // online softmax, 16-lane row reductions
        float alpha[8];
#pragma unroll
        for (int r = 0; r < 8; r++) {
            float mx = fmaxf(sc[0][r], sc[1][r]);
            mx = fmaxf(mx, __shfl_xor(mx, 1, 32));
            mx = fmaxf(mx, __shfl_xor(mx, 2, 32));
            mx = fmaxf(mx, __shfl_xor(mx, 4, 32));
            mx = fmaxf(mx, __shfl_xor(mx, 8, 32));
            float mnew = fmaxf(mstate[r], mx);
            alpha[r] = __expf(mstate[r] - mnew);
            mstate[r] = mnew;
            float p0 = __expf(sc[0][r] - mnew);
            float p1 = __expf(sc[1][r] - mnew);
            sc[0][r] = p0; sc[1][r] = p1;
            float ps = p0 + p1;
            ps += __shfl_xor(ps, 1, 32);
            ps += __shfl_xor(ps, 2, 32);
            ps += __shfl_xor(ps, 4, 32);
            ps += __shfl_xor(ps, 8, 32);
            lstate[r] = lstate[r] * alpha[r] + ps;
        }
#pragma unroll
        for (int t = 0; t < 4; t++)
#pragma unroll
            for (int r = 0; r < 8; r++) o[t][r] *= alpha[r];
        // C-layout P -> LDS (bf16) for A-layout reload
#pragma unroll
        for (int r = 0; r < 8; r++) {
            p_lds[(hi * 8 + r) * 32 + (lane & 15)]      = f2bf(sc[0][r]);
            p_lds[(hi * 8 + r) * 32 + 16 + (lane & 15)] = f2bf(sc[1][r]);
        }
        // stage V chunk transposed: vt[dh][key]
        {
            int j = j0 + lane;
            const unsigned short* vr = key_row(vb, pv, h, w, j, S);
#pragma unroll
            for (int d8 = 0; d8 < 8; d8++) {
                uint4 d = *(const uint4*)(vr + d8 * 8);
                const unsigned short* dd = (const unsigned short*)&d;
#pragma unroll
                for (int e = 0; e < 8; e++) vt[(d8 * 8 + e) * 32 + lane] = dd[e];
            }
        }
        __syncthreads();
        v16bf pf = frag_a_from_row(&p_lds[(lane & 15) * 32], 0, lane);
#pragma unroll
        for (int nt = 0; nt < 4; nt++) {
            v16bf vf = frag_b_from_row(&vt[(nt * 16 + (lane & 15)) * 32], 0, lane);
            o[nt] = wmma_bf16(pf, vf, o[nt]);
        }
        __syncthreads();
    }
    // normalize + store to [S][H*64] bf16 for the out-proj GEMM
#pragma unroll
    for (int nt = 0; nt < 4; nt++) {
#pragma unroll
        for (int r = 0; r < 8; r++) {
            int i = qt * 16 + hi * 8 + r;
            if (i < SEG_) {
                float v = o[nt][r] / fmaxf(lstate[r], 1e-20f);
                size_t s = (size_t)w * SEG_ + i;
                outb[s * HD_ + h * DH_ + nt * 16 + (lane & 15)] = f2bf(v);
            }
        }
    }
}

// ---------------- host orchestration ----------------
extern "C" void kernel_launch(void* const* d_in, const int* in_sizes, int n_in,
                              void* d_out, int out_size, void* d_ws, size_t ws_size,
                              hipStream_t stream) {
    (void)in_sizes; (void)n_in; (void)out_size; (void)ws_size;
    const float* hidden  = (const float*)d_in[0];
    const float* in_ln   = (const float*)d_in[1];
    const float* post_ln = (const float*)d_in[2];
    const float* attn_ln = (const float*)d_in[3];
    const float* w_qkv   = (const float*)d_in[4];
    const float* w_out   = (const float*)d_in[5];
    const float* pm      = (const float*)d_in[6];
    const float* w_gate  = (const float*)d_in[7];
    const float* w_up    = (const float*)d_in[8];
    const float* w_down  = (const float*)d_in[9];

    char* wsp = (char*)d_ws;
    size_t off = 0;
    auto carve = [&](size_t bytes) -> void* {
        void* p = wsp + off;
        off += (bytes + 255) & ~(size_t)255;
        return p;
    };
    unsigned short* wqkv_b  = (unsigned short*)carve((size_t)D_ * QN_ * 2);
    unsigned short* wout_b  = (unsigned short*)carve((size_t)HD_ * D_ * 2);
    unsigned short* wgate_b = (unsigned short*)carve((size_t)D_ * FF_ * 2);
    unsigned short* wup_b   = (unsigned short*)carve((size_t)D_ * FF_ * 2);
    unsigned short* wdown_b = (unsigned short*)carve((size_t)FF_ * D_ * 2);
    unsigned short* pkv_b   = (unsigned short*)carve((size_t)2 * H_ * P_ * DH_ * 2);
    unsigned short* xn_b    = (unsigned short*)carve((size_t)S_ * D_ * 2);
    float*          qkv_f   = (float*)carve((size_t)S_ * QN_ * 4);
    unsigned short* q_b     = (unsigned short*)carve((size_t)H_ * S_ * DH_ * 2);
    unsigned short* k_b     = (unsigned short*)carve((size_t)H_ * S_ * DH_ * 2);
    unsigned short* v_b     = (unsigned short*)carve((size_t)H_ * S_ * DH_ * 2);
    unsigned short* ao_b    = (unsigned short*)carve((size_t)S_ * HD_ * 2);
    float*          x_f     = (float*)carve((size_t)S_ * D_ * 4);
    unsigned short* h2_b    = (unsigned short*)carve((size_t)S_ * D_ * 2);
    unsigned short* act_b   = (unsigned short*)carve((size_t)S_ * FF_ * 2);

    // 1) weight + persistent-memory conversion to bf16
    cvt_bf16_kernel<<<2048, 256, 0, stream>>>(w_qkv, wqkv_b, (size_t)D_ * QN_);
    cvt_bf16_kernel<<<2048, 256, 0, stream>>>(w_out, wout_b, (size_t)HD_ * D_);
    cvt_bf16_kernel<<<2048, 256, 0, stream>>>(w_gate, wgate_b, (size_t)D_ * FF_);
    cvt_bf16_kernel<<<2048, 256, 0, stream>>>(w_up, wup_b, (size_t)D_ * FF_);
    cvt_bf16_kernel<<<2048, 256, 0, stream>>>(w_down, wdown_b, (size_t)FF_ * D_);
    cvt_bf16_kernel<<<64, 256, 0, stream>>>(pm, pkv_b, (size_t)2 * H_ * P_ * DH_);

    // 2) xn = rms(rms(hidden, input_ln), attn_norm)  [bf16]
    rmsnorm2_kernel<<<S_, 256, 0, stream>>>(hidden, in_ln, attn_ln, xn_b, D_);

    // 3) qkv = xn @ w_qkv   [fp32]
    gemm_bf16_kernel<0, 128><<<dim3(QN_ / 128, (S_ + BM - 1) / BM), 256, 0, stream>>>(
        xn_b, wqkv_b, nullptr, qkv_f, nullptr, nullptr, S_, QN_, D_);

    // 4) rotary + scale + repack
    rotary_kernel<<<S_, 256, 0, stream>>>(qkv_f, q_b, k_b, v_b, S_);

    // 5) windowed flash attention
    attn_kernel<<<H_ * W_ * QT_, 32, 0, stream>>>(
        q_b, k_b, v_b, pkv_b, pkv_b + (size_t)H_ * P_ * DH_, ao_b, S_);

    // 6) x = hidden + attn_out @ w_out   [fp32]
    gemm_bf16_kernel<1, 128><<<dim3(D_ / 128, (S_ + BM - 1) / BM), 256, 0, stream>>>(
        ao_b, wout_b, nullptr, x_f, nullptr, hidden, S_, D_, HD_);

    // 7) h2 = rms(x, post_ln)   [bf16]
    rmsnorm1_kernel<<<S_, 256, 0, stream>>>(x_f, post_ln, h2_b, D_);

    // 8) act = silu(h2 @ w_gate) * (h2 @ w_up)   [bf16, fused dual GEMM]
    gemm_bf16_kernel<2, 64><<<dim3(FF_ / 64, (S_ + BM - 1) / BM), 256, 0, stream>>>(
        h2_b, wgate_b, wup_b, nullptr, act_b, nullptr, S_, FF_, D_);

    // 9) out = x + act @ w_down   [fp32]
    gemm_bf16_kernel<1, 128><<<dim3(D_ / 128, (S_ + BM - 1) / BM), 256, 0, stream>>>(
        act_b, wdown_b, nullptr, (float*)d_out, nullptr, x_f, S_, D_, FF_);
}
